// Qwen3Attention_44538810860124
// MI455X (gfx1250) — matrix operand order repaired and checked by reading, not compiled, not run on hardware
//
#include <hip/hip_runtime.h>
#include <hip/hip_bf16.h>

typedef __bf16 v16bf __attribute__((ext_vector_type(16)));
typedef float  v8f   __attribute__((ext_vector_type(8)));

union Frag {
    v16bf bf;
    uint4 u4[2];
};

__device__ __forceinline__ unsigned short f2bf(float x) {
    unsigned int u = __float_as_uint(x);
    unsigned int r = (u + 0x7FFFu + ((u >> 16) & 1u)) >> 16;
    return (unsigned short)r;
}

__device__ __forceinline__ v8f wmma_bf16(const Frag& a, const Frag& b, v8f c) {
    return __builtin_amdgcn_wmma_f32_16x16x32_bf16(
        false, a.bf, false, b.bf, (short)0, c, false, false);
}

// CDNA5 async global->LDS copy (16B per lane), tracked by ASYNCcnt.
__device__ __forceinline__ void async_ld_b128(unsigned int lds_addr, const void* gaddr) {
    asm volatile("global_load_async_to_lds_b128 %0, %1, off"
                 :: "v"(lds_addr), "v"(gaddr) : "memory");
}
__device__ __forceinline__ unsigned int lds_off(const void* p) {
    return (unsigned int)(uintptr_t)p;  // low 32 bits of flat shared addr = LDS offset
}

// ---------------------------------------------------------------------------
// fp32 -> bf16 cast (grid-stride)
// ---------------------------------------------------------------------------
__global__ void cast_f32_bf16(const float* __restrict__ src,
                              unsigned short* __restrict__ dst, int n) {
    int i = blockIdx.x * blockDim.x + threadIdx.x;
    int stride = gridDim.x * blockDim.x;
    for (; i < n; i += stride) dst[i] = f2bf(src[i]);
}

// ---------------------------------------------------------------------------
// NT GEMM: C[M,N] = A[M,K] (bf16 row-major) * W[N,K]^T (bf16 row-major)
// block 256 = 8 waves; block tile 128x256; wave tile 64x64; K step 32.
// Double-buffered LDS filled with global_load_async_to_lds_b128.
// ---------------------------------------------------------------------------
__global__ __launch_bounds__(256) void gemm_nt_bf16(
    const unsigned short* __restrict__ A, int lda,
    const unsigned short* __restrict__ W, int ldw,
    float* __restrict__ C, int ldc, int K) {
    __shared__ __align__(16) unsigned short As[2][128 * 40];
    __shared__ __align__(16) unsigned short Ws[2][256 * 40];

    const int tid  = threadIdx.x;
    const int lane = tid & 31;
    const int wid  = tid >> 5;
    const int bm   = blockIdx.y;
    const int bn   = blockIdx.x;
    const int wm   = (wid & 1) * 64;   // wave M offset in block tile
    const int wn   = (wid >> 1) * 64;  // wave N offset in block tile
    const int ln   = lane & 15;
    const int lh   = lane >> 4;

    v8f acc[4][4];
    for (int i = 0; i < 4; i++)
        for (int j = 0; j < 4; j++)
            for (int v = 0; v < 8; v++) acc[i][j][v] = 0.0f;

    // staging coords: A -> 2 x b128/thread, W -> 4 x b128/thread (one row)
    const int ra = tid >> 1, ca = (tid & 1) * 16;
    const unsigned int asBase = lds_off(&As[0][0]);
    const unsigned int wsBase = lds_off(&Ws[0][0]);
    const unsigned short* gA = A + (size_t)(bm * 128 + ra) * lda + ca;
    const unsigned short* gW = W + (size_t)(bn * 256 + tid) * ldw;

    auto stage = [&](int buf, int ko) {
        const unsigned short* ga = gA + ko;
        unsigned int la = asBase + (unsigned)buf * (128 * 40 * 2) + (ra * 40 + ca) * 2;
        async_ld_b128(la,      ga);
        async_ld_b128(la + 16, ga + 8);
        const unsigned short* gw = gW + ko;
        unsigned int lw = wsBase + (unsigned)buf * (256 * 40 * 2) + (tid * 40) * 2;
        async_ld_b128(lw,      gw);
        async_ld_b128(lw + 16, gw + 8);
        async_ld_b128(lw + 32, gw + 16);
        async_ld_b128(lw + 48, gw + 24);
    };

    stage(0, 0);
    int buf = 0;
    for (int ko = 0; ko < K; ko += 32, buf ^= 1) {
        if (ko + 32 < K) {
            stage(buf ^ 1, ko + 32);
            asm volatile("s_wait_asynccnt 6" ::: "memory");  // prev 6 done (in-order)
        } else {
            asm volatile("s_wait_asynccnt 0" ::: "memory");
        }
        __syncthreads();

        const unsigned short* as = &As[buf][0];
        const unsigned short* wsb = &Ws[buf][0];
        Frag bfr[4];
#pragma unroll
        for (int j = 0; j < 4; j++) {
            const uint4* p = (const uint4*)(wsb + (wn + j * 16 + ln) * 40 + lh * 16);
            bfr[j].u4[0] = p[0]; bfr[j].u4[1] = p[1];
        }
#pragma unroll
        for (int i = 0; i < 4; i++) {
            Frag af;
            const unsigned short* pa = as + (wm + i * 16 + ln) * 40 + lh * 16;
            af.u4[0] = *(const uint4*)pa;
            af.u4[1] = *(const uint4*)(pa + 8);
#pragma unroll
            for (int j = 0; j < 4; j++)
                acc[i][j] = wmma_bf16(af, bfr[j], acc[i][j]);
        }
        __syncthreads();
    }

#pragma unroll
    for (int i = 0; i < 4; i++)
#pragma unroll
        for (int j = 0; j < 4; j++)
#pragma unroll
            for (int v = 0; v < 8; v++) {
                int row = bm * 128 + wm + i * 16 + v + 8 * lh;
                int col = bn * 256 + wn + j * 16 + ln;
                C[(size_t)row * ldc + col] = acc[i][j][v];
            }
}

// ---------------------------------------------------------------------------
// Fused per-head RMSNorm + RoPE + bf16 pack. One wave32 per 128-wide row.
// Row sections: [0,65536) Q rows, [65536,81920) K rows, [81920,98304) V rows.
// ---------------------------------------------------------------------------
__global__ __launch_bounds__(256) void normrope(
    const float* __restrict__ Qf, const float* __restrict__ Kf,
    const float* __restrict__ Vf, const int* __restrict__ pos_ids,
    const float* __restrict__ qw, const float* __restrict__ kw,
    unsigned short* __restrict__ Qb, unsigned short* __restrict__ Kb,
    unsigned short* __restrict__ Vb) {
    const int wid  = threadIdx.x >> 5;
    const int lane = threadIdx.x & 31;
    const int ridx = blockIdx.x * 8 + wid;

    const float* src;
    const float* wgt = nullptr;
    unsigned short* dst;
    int pos = 0;
    bool rope = false;

    if (ridx < 65536) {                 // Q
        int h = ridx & 31, s = ridx >> 5;
        src = Qf + (size_t)s * 4096 + h * 128;
        wgt = qw; rope = true; pos = pos_ids[s];
        dst = Qb + ((size_t)h * 2048 + s) * 128;
    } else if (ridx < 81920) {          // K
        int rr = ridx - 65536;
        int h = rr & 7, s = rr >> 3;
        src = Kf + (size_t)s * 1024 + h * 128;
        wgt = kw; rope = true; pos = pos_ids[s];
        dst = Kb + ((size_t)h * 2048 + s) * 128;
    } else {                            // V
        int rr = ridx - 81920;
        int h = rr & 7, s = rr >> 3;
        src = Vf + (size_t)s * 1024 + h * 128;
        dst = Vb + ((size_t)h * 2048 + s) * 128;
    }

    float x0 = src[lane], x1 = src[lane + 32], x2 = src[lane + 64], x3 = src[lane + 96];

    if (!rope) {  // branch is wave-uniform
        dst[lane]      = f2bf(x0);
        dst[lane + 32] = f2bf(x1);
        dst[lane + 64] = f2bf(x2);
        dst[lane + 96] = f2bf(x3);
        return;
    }

    float ss = x0 * x0 + x1 * x1 + x2 * x2 + x3 * x3;
    for (int off = 16; off; off >>= 1) ss += __shfl_xor(ss, off);
    float sc = rsqrtf(ss * (1.0f / 128.0f) + 1e-6f);
    x0 *= sc * wgt[lane];
    x1 *= sc * wgt[lane + 32];
    x2 *= sc * wgt[lane + 64];
    x3 *= sc * wgt[lane + 96];

    // RoPE: d pairs (lane, lane+64) and (lane+32, lane+96); ln(1e6)/64
    const float kfreq = 0.21586735246819178f;
    float p  = (float)pos;
    float a0 = p * __expf(-kfreq * (float)lane);
    float a1 = p * __expf(-kfreq * (float)(lane + 32));
    float c0 = __cosf(a0), s0 = __sinf(a0);
    float c1 = __cosf(a1), s1 = __sinf(a1);

    dst[lane]      = f2bf(x0 * c0 - x2 * s0);
    dst[lane + 32] = f2bf(x1 * c1 - x3 * s1);
    dst[lane + 64] = f2bf(x2 * c0 + x0 * s0);
    dst[lane + 96] = f2bf(x3 * c1 + x1 * s1);
}

// ---------------------------------------------------------------------------
// Causal flash attention. grid = (S/64, n_heads), block = 128 (4 waves).
// Wave w owns 16 query rows; per 64-key block: S = Q K^T (16 WMMA/wave),
// fp32 online softmax, P bf16 overlay, O += P V (16 WMMA/wave).
// ---------------------------------------------------------------------------
__global__ __launch_bounds__(128) void attn_kernel(
    const unsigned short* __restrict__ Qb,
    const unsigned short* __restrict__ Kb,
    const unsigned short* __restrict__ Vb,
    unsigned short* __restrict__ Ob) {
    __shared__ __align__(16) unsigned short Kt[64 * 136];   // K tile [k][d]
    __shared__ __align__(16) unsigned short VT[128 * 72];   // V tile transposed [d][k]
    __shared__ __align__(16) float Ssc[4][16][68];          // score strip (P overlays)
    __shared__ float rowfac[4][16];

    const int h   = blockIdx.y;
    const int qb  = blockIdx.x;
    const int tid = threadIdx.x;
    const int lane = tid & 31;
    const int w    = tid >> 5;
    const int hk   = h >> 2;                  // GQA: 4 Q heads per KV head
    const int ln   = lane & 15;
    const int lh   = lane >> 4;
    const unsigned int ktBase = lds_off(&Kt[0]);

    // Q fragments for this wave's 16 rows (registers, reused for all k-blocks)
    Frag qf[4];
    {
        const unsigned short* qrow =
            Qb + ((size_t)h * 2048 + qb * 64 + w * 16 + ln) * 128;
#pragma unroll
        for (int d = 0; d < 4; d++) {
            const unsigned short* p = qrow + d * 32 + lh * 16;
            qf[d].u4[0] = *(const uint4*)p;
            qf[d].u4[1] = *(const uint4*)(p + 8);
        }
    }

    v8f oacc[8];
    for (int dt = 0; dt < 8; dt++)
        for (int v = 0; v < 8; v++) oacc[dt][v] = 0.0f;
    float m_i = -3.0e38f, l_i = 0.0f;
    const float scaling = 0.08838834764831845f;  // 1/sqrt(128)
    const int rowbase = qb * 64 + w * 16;

    for (int kb = 0; kb <= qb; kb++) {
        // K tile: straight copy -> async global->LDS
        for (int c = tid; c < 1024; c += 128) {
            int r = c >> 4, c8 = (c & 15) << 3;
            async_ld_b128(ktBase + (r * 136 + c8) * 2,
                          Kb + ((size_t)hk * 2048 + kb * 64 + r) * 128 + c8);
        }
        // V tile: transpose through VGPRs
        for (int c = tid; c < 1024; c += 128) {
            int r = c >> 4, c8 = (c & 15) << 3;
            uint4 tv = *(const uint4*)(Vb + ((size_t)hk * 2048 + kb * 64 + r) * 128 + c8);
            const unsigned short* ts = (const unsigned short*)&tv;
#pragma unroll
            for (int j = 0; j < 8; j++) VT[(c8 + j) * 72 + r] = ts[j];
        }
        asm volatile("s_wait_asynccnt 0" ::: "memory");
        __syncthreads();

        // S = Q K^T (16x64 strip per wave)
        v8f sacc[4];
        for (int nt = 0; nt < 4; nt++)
            for (int v = 0; v < 8; v++) sacc[nt][v] = 0.0f;
#pragma unroll
        for (int d = 0; d < 4; d++) {
#pragma unroll
            for (int nt = 0; nt < 4; nt++) {
                Frag bf;
                const unsigned short* p = Kt + (nt * 16 + ln) * 136 + d * 32 + lh * 16;
                bf.u4[0] = *(const uint4*)p;
                bf.u4[1] = *(const uint4*)(p + 8);
                sacc[nt] = wmma_bf16(qf[d], bf, sacc[nt]);
            }
        }

        // scale + causal mask, write score strip to LDS
#pragma unroll
        for (int nt = 0; nt < 4; nt++) {
            int col = kb * 64 + nt * 16 + ln;
#pragma unroll
            for (int v = 0; v < 8; v++) {
                int row = rowbase + v + 8 * lh;
                float val = sacc[nt][v] * scaling;
                if (col > row) val = -1.0e9f;
                Ssc[w][v + 8 * lh][nt * 16 + ln] = val;
            }
        }

        // online softmax: lane r<16 owns row r of this wave's strip
        if (lane < 16) {
            float* srow = &Ssc[w][lane][0];
            float mnew = m_i;
            for (int j = 0; j < 64; j++) mnew = fmaxf(mnew, srow[j]);
            unsigned short* prow = (unsigned short*)srow;  // bf16 overlay (writes trail reads)
            float sum = 0.0f;
            for (int j = 0; j < 64; j++) {
                float pj = __expf(srow[j] - mnew);
                sum += pj;
                prow[j] = f2bf(pj);
            }
            float fac = __expf(m_i - mnew);
            l_i = l_i * fac + sum;
            m_i = mnew;
            rowfac[w][lane] = fac;
        }

        // rescale running O by per-row factor
#pragma unroll
        for (int dt = 0; dt < 8; dt++)
#pragma unroll
            for (int v = 0; v < 8; v++)
                oacc[dt][v] *= rowfac[w][v + 8 * lh];

        // O += P V
#pragma unroll
        for (int kc = 0; kc < 2; kc++) {
            Frag pf;
            const unsigned short* pp = (const unsigned short*)&Ssc[w][ln][0];
            pf.u4[0] = *(const uint4*)(pp + kc * 32 + lh * 16);
            pf.u4[1] = *(const uint4*)(pp + kc * 32 + 8 + lh * 16);
#pragma unroll
            for (int dt = 0; dt < 8; dt++) {
                Frag bv;
                const unsigned short* pv = VT + (dt * 16 + ln) * 72 + kc * 32 + lh * 16;
                bv.u4[0] = *(const uint4*)pv;
                bv.u4[1] = *(const uint4*)(pv + 8);
                oacc[dt] = wmma_bf16(pf, bv, oacc[dt]);
            }
        }
        __syncthreads();
    }

    // normalize and store bf16 O in [s][h*128+d] layout for the Wo GEMM
    if (lane < 16) rowfac[w][lane] = 1.0f / l_i;
#pragma unroll
    for (int dt = 0; dt < 8; dt++)
#pragma unroll
        for (int v = 0; v < 8; v++) {
            float o = oacc[dt][v] * rowfac[w][v + 8 * lh];
            int s = rowbase + v + 8 * lh;
            Ob[(size_t)s * 4096 + h * 128 + dt * 16 + ln] = f2bf(o);
        }
}

// ---------------------------------------------------------------------------
// host launcher
// ---------------------------------------------------------------------------
extern "C" void kernel_launch(void* const* d_in, const int* in_sizes, int n_in,
                              void* d_out, int out_size, void* d_ws, size_t ws_size,
                              hipStream_t stream) {
    (void)in_sizes; (void)n_in; (void)out_size; (void)ws_size;

    const float* hs  = (const float*)d_in[0];
    const int*   pid = (const int*)d_in[1];
    const float* Wq  = (const float*)d_in[2];
    const float* Wk  = (const float*)d_in[3];
    const float* Wv  = (const float*)d_in[4];
    const float* Wo  = (const float*)d_in[5];
    const float* qnw = (const float*)d_in[6];
    const float* knw = (const float*)d_in[7];
    float* out = (float*)d_out;

    char* ws = (char*)d_ws;
    size_t off = 0;
    auto alloc = [&](size_t bytes) {
        size_t o = off;
        off += (bytes + 255) & ~(size_t)255;
        return o;
    };
    const size_t S = 2048, H = 4096, DK = 1024;
    size_t o_hsb = alloc(S * H * 2);       // hs bf16
    size_t o_wqb = alloc(H * H * 2);       // Wq bf16  (reused for O bf16 later)
    size_t o_wkb = alloc(DK * H * 2);
    size_t o_wvb = alloc(DK * H * 2);
    size_t o_wob = alloc(H * H * 2);
    size_t o_qf  = alloc(S * H * 4);       // Q fp32
    size_t o_kf  = alloc(S * DK * 4);
    size_t o_vf  = alloc(S * DK * 4);
    size_t o_qb  = alloc(S * H * 2);       // Q bf16 post rope, [h][s][d]
    size_t o_kb  = alloc(S * DK * 2);
    size_t o_vb  = alloc(S * DK * 2);

    unsigned short* hsb = (unsigned short*)(ws + o_hsb);
    unsigned short* wqb = (unsigned short*)(ws + o_wqb);
    unsigned short* wkb = (unsigned short*)(ws + o_wkb);
    unsigned short* wvb = (unsigned short*)(ws + o_wvb);
    unsigned short* wob = (unsigned short*)(ws + o_wob);
    float* qf = (float*)(ws + o_qf);
    float* kf = (float*)(ws + o_kf);
    float* vf = (float*)(ws + o_vf);
    unsigned short* qb = (unsigned short*)(ws + o_qb);
    unsigned short* kb = (unsigned short*)(ws + o_kb);
    unsigned short* vb = (unsigned short*)(ws + o_vb);
    unsigned short* ob = wqb;  // Wq bf16 no longer needed once attention runs

    // 1) casts
    cast_f32_bf16<<<4096, 256, 0, stream>>>(hs, hsb, (int)(S * H));
    cast_f32_bf16<<<4096, 256, 0, stream>>>(Wq, wqb, (int)(H * H));
    cast_f32_bf16<<<4096, 256, 0, stream>>>(Wk, wkb, (int)(DK * H));
    cast_f32_bf16<<<4096, 256, 0, stream>>>(Wv, wvb, (int)(DK * H));
    cast_f32_bf16<<<4096, 256, 0, stream>>>(Wo, wob, (int)(H * H));

    // 2) QKV projections (NT GEMMs), block tile 128x256
    gemm_nt_bf16<<<dim3(16, 16), 256, 0, stream>>>(hsb, 4096, wqb, 4096, qf, 4096, 4096);
    gemm_nt_bf16<<<dim3(4, 16), 256, 0, stream>>>(hsb, 4096, wkb, 4096, kf, 1024, 4096);
    gemm_nt_bf16<<<dim3(4, 16), 256, 0, stream>>>(hsb, 4096, wvb, 4096, vf, 1024, 4096);

    // 3) RMSNorm + RoPE + bf16 pack (98304 rows, 8 rows/block)
    normrope<<<12288, 256, 0, stream>>>(qf, kf, vf, pid, qnw, knw, qb, kb, vb);

    // 4) causal flash attention
    attn_kernel<<<dim3(32, 32), 128, 0, stream>>>(qb, kb, vb, ob);

    // 5) output projection -> fp32 d_out
    gemm_nt_bf16<<<dim3(16, 16), 256, 0, stream>>>(ob, 4096, wob, 4096, out, 4096, 4096);
}
